// AttentionBlock_51049981281013
// MI455X (gfx1250) — matrix operand order repaired and checked by reading, not compiled, not run on hardware
//
#include <hip/hip_runtime.h>

// AttentionBlock for MI455X (gfx1250, wave32, WMMA, async global->LDS).
// B=8, C=512, H=W=64 -> T=4096, G=32, EPS=1e-5.
// All GEMMs via v_wmma_f32_16x16x32_bf16, 128x128 macro tiles, double-buffered LDS.
// Workspace use: ~118 MB.

#define BATCH 8
#define CCH   512
#define TT    4096
#define NGRP  32
#define QK_SCALE 0.21022410381342863f   // 512^-0.25

typedef __attribute__((ext_vector_type(16))) __bf16 v16bf;
typedef __attribute__((ext_vector_type(8)))  float  v8f;
typedef __attribute__((ext_vector_type(4)))  unsigned int v4u;
typedef __attribute__((ext_vector_type(4)))  int v4i_t;

union Frag16 { v16bf v; unsigned int u[8]; };

static __device__ inline unsigned short f2bf(float f) {
    union { float f; unsigned int u; } x; x.f = f;
    unsigned int r = (x.u + 0x7FFFu + ((x.u >> 16) & 1u)) >> 16;   // RNE
    return (unsigned short)r;
}

// ---- CDNA5 async global->LDS path (guarded: falls back to sync copies) ----
#if defined(__gfx1250__) && \
    __has_builtin(__builtin_amdgcn_global_load_async_to_lds_b128) && \
    __has_builtin(__builtin_amdgcn_s_wait_asynccnt)
#define USE_ASYNC 1
#else
#define USE_ASYNC 0
#endif

#if USE_ASYNC
// builtin signature (from clang diagnostics): arg0 = v4i addrspace(1)* (global),
// arg1 = v4i addrspace(3)* (LDS), arg2 = imm offset, arg3 = imm cpol
typedef __attribute__((address_space(1))) v4i_t* gv4i_p;
typedef __attribute__((address_space(3))) v4i_t* lv4i_p;
static __device__ inline gv4i_p gcast(const void* p) {
    return (gv4i_p)(unsigned long long)p;
}
static __device__ inline lv4i_p lcast(void* p) {
    // generic LDS address: low 32 bits are the LDS offset (aperture in high bits)
    return (lv4i_p)(unsigned int)(unsigned long long)p;
}
#endif

// ---------------------------------------------------------------------------
// Kernel: f32 -> bf16 (pre-convert GEMM weight operands once)
// ---------------------------------------------------------------------------
__global__ void f32_to_bf16_kernel(const float* __restrict__ src,
                                   unsigned short* __restrict__ dst, int n) {
    int i = blockIdx.x * 256 + threadIdx.x;
    if (i < n) dst[i] = f2bf(src[i]);
}

// ---------------------------------------------------------------------------
// Kernel: GroupNorm statistics. One block per (b,g); group channels are
// contiguous: 16 channels * 4096 = 65536 contiguous floats.
// ---------------------------------------------------------------------------
__global__ void gn_stats_kernel(const float* __restrict__ x, float* __restrict__ stats) {
    const int bg = blockIdx.x;
    const float* base = x + (size_t)bg * (16 * TT);
    float s = 0.f, ss = 0.f;
    for (int i = threadIdx.x; i < 16 * TT; i += 256) {
        float v = base[i];
        s += v; ss += v * v;
    }
    __shared__ float rs[256], rss[256];
    rs[threadIdx.x] = s; rss[threadIdx.x] = ss;
    __syncthreads();
    for (int off = 128; off > 0; off >>= 1) {
        if (threadIdx.x < off) {
            rs[threadIdx.x]  += rs[threadIdx.x + off];
            rss[threadIdx.x] += rss[threadIdx.x + off];
        }
        __syncthreads();
    }
    if (threadIdx.x == 0) {
        const float inv = 1.0f / (16.0f * TT);
        float mu  = rs[0] * inv;
        float var = rss[0] * inv - mu * mu;
        stats[2 * bg]     = mu;
        stats[2 * bg + 1] = rsqrtf(var + 1e-5f);
    }
}

// ---------------------------------------------------------------------------
// Kernel: normalize + affine + cast to bf16, transposed to [T][C].
// ---------------------------------------------------------------------------
__global__ void gn_apply_kernel(const float* __restrict__ xb,
                                const float* __restrict__ stats_b,
                                const float* __restrict__ gnw,
                                const float* __restrict__ gnb,
                                unsigned short* __restrict__ h) {
    int idx = blockIdx.x * 256 + threadIdx.x;      // idx = t*C + c
    int t = idx >> 9;
    int c = idx & (CCH - 1);
    int g = c >> 4;
    float mu = stats_b[2 * g], rstd = stats_b[2 * g + 1];
    float v = (xb[(size_t)c * TT + t] - mu) * rstd * gnw[c] + gnb[c];
    h[idx] = f2bf(v);
}

// ---------------------------------------------------------------------------
// Kernel: row softmax over attn [4096][4096] f32 -> bf16 probabilities.
// ---------------------------------------------------------------------------
__global__ void softmax_kernel(const float* __restrict__ attn,
                               unsigned short* __restrict__ attn_bf) {
    const int t = blockIdx.x;
    const float* row = attn + (size_t)t * TT;
    __shared__ float red[256];
    float m = -1e30f;
    for (int s = threadIdx.x; s < TT; s += 256) m = fmaxf(m, row[s]);
    red[threadIdx.x] = m; __syncthreads();
    for (int off = 128; off > 0; off >>= 1) {
        if (threadIdx.x < off) red[threadIdx.x] = fmaxf(red[threadIdx.x], red[threadIdx.x + off]);
        __syncthreads();
    }
    m = red[0];
    __syncthreads();
    float sum = 0.f;
    for (int s = threadIdx.x; s < TT; s += 256) sum += __expf(row[s] - m);
    red[threadIdx.x] = sum; __syncthreads();
    for (int off = 128; off > 0; off >>= 1) {
        if (threadIdx.x < off) red[threadIdx.x] += red[threadIdx.x + off];
        __syncthreads();
    }
    float inv = 1.0f / red[0];
    unsigned short* orow = attn_bf + (size_t)t * TT;
    for (int s = threadIdx.x; s < TT; s += 256)
        orow[s] = f2bf(__expf(row[s] - m) * inv);
}

// ---------------------------------------------------------------------------
// 128x128-macro-tile bf16 WMMA GEMM, 256 threads (8 wave32), K step 32,
// double-buffered LDS with (optionally) async global->LDS staging.
//   D(M,N) = sum_K A(M,K) * B(K,N)
// A: bf16 row-major [M][K]. B: "N-major": (k,n) at Bt[n*ldb + k].
// Wave grid 4x2; each wave computes 32(M) x 64(N): acc[2][4].
// Epilogues:
//   EPI=0 QKV  : +qkv_bias; q,k *= QK_SCALE; q,k -> [T][C] bf16, v -> [C][T] bf16
//   EPI=1 SCOR : f32 store, ld = 4096
//   EPI=2 AV   : bf16 store [T][C]
//   EPI=3 PROJ : f32 store + proj bias + residual x, ld = 4096
// ---------------------------------------------------------------------------
template <int EPI>
__global__ void gemm128_wmma(const unsigned short* __restrict__ Abf, int lda,
                             const unsigned short* __restrict__ Bt, int ldb,
                             int K,
                             void* out0, void* out1, void* out2,
                             const float* __restrict__ bias,
                             const float* __restrict__ resid) {
    // [2 buffers][128 rows][32 bf16 = 16 dwords]
    __shared__ unsigned int sA[2 * 2048];
    __shared__ unsigned int sB[2 * 2048];

    const int m0 = blockIdx.x * 128;
    const int n0 = blockIdx.y * 128;

    const int tid  = threadIdx.x;
    const int lane = tid & 31;
    const int wid  = tid >> 5;
    const int wm   = (wid & 3) * 32;       // 4 waves along M
    const int wn   = (wid >> 2) * 64;      // 2 waves along N
    const int lr   = lane & 15;
    const int lhi  = lane >> 4;

    v8f zero = {0.f, 0.f, 0.f, 0.f, 0.f, 0.f, 0.f, 0.f};
    v8f acc[2][4] = {{zero, zero, zero, zero}, {zero, zero, zero, zero}};

    const int nk = K >> 5;

    // ---- tile staging: 512 x b128 per operand, 2 per thread per operand ----
    auto stage = [&](int buf, int k0) {
        unsigned int* sAb = sA + buf * 2048;
        unsigned int* sBb = sB + buf * 2048;
        for (int i = tid; i < 512; i += 256) {
            int row = i >> 2, quad = i & 3;
            const unsigned short* aS = Abf + (size_t)(m0 + row) * lda + k0 + quad * 8;
            const unsigned short* bS = Bt  + (size_t)(n0 + row) * ldb + k0 + quad * 8;
            unsigned int* aD = sAb + row * 16 + quad * 4;
            unsigned int* bD = sBb + row * 16 + quad * 4;
#if USE_ASYNC
            __builtin_amdgcn_global_load_async_to_lds_b128(gcast(aS), lcast(aD), 0, 0);
            __builtin_amdgcn_global_load_async_to_lds_b128(gcast(bS), lcast(bD), 0, 0);
#else
            *(v4u*)aD = *(const v4u*)aS;
            *(v4u*)bD = *(const v4u*)bS;
#endif
        }
    };

    stage(0, 0);
    for (int kb = 0; kb < nk; ++kb) {
        const int cur = kb & 1;
        if (kb + 1 < nk) {
            stage(cur ^ 1, (kb + 1) << 5);   // prefetch next tile
#if USE_ASYNC
            // current tile's 4 asyncs done when only next tile's 4 remain
            __builtin_amdgcn_s_wait_asynccnt(4);
#endif
        } else {
#if USE_ASYNC
            __builtin_amdgcn_s_wait_asynccnt(0);
#endif
        }
        __syncthreads();

        const unsigned int* cA = sA + cur * 2048;
        const unsigned int* cB = sB + cur * 2048;

        Frag16 fa[2], fb[4];
#pragma unroll
        for (int mi = 0; mi < 2; ++mi)
#pragma unroll
            for (int j = 0; j < 8; ++j)
                fa[mi].u[j] = cA[(wm + mi * 16 + lr) * 16 +
                                 ((j >> 2) << 3) + (lhi << 2) + (j & 3)];
#pragma unroll
        for (int ni = 0; ni < 4; ++ni)
#pragma unroll
            for (int j = 0; j < 8; ++j)
                fb[ni].u[j] = cB[(wn + ni * 16 + lr) * 16 +
                                 ((j >> 2) << 3) + (lhi << 2) + (j & 3)];

#pragma unroll
        for (int mi = 0; mi < 2; ++mi)
#pragma unroll
            for (int ni = 0; ni < 4; ++ni)
                acc[mi][ni] = __builtin_amdgcn_wmma_f32_16x16x32_bf16(
                    false, fa[mi].v, false, fb[ni].v,
                    (short)0, acc[mi][ni], false, false);
        __syncthreads();
    }

    // ---- epilogue: D layout lane = N, dword j = M (j / 8+j per lane half) ----
#pragma unroll
    for (int mi = 0; mi < 2; ++mi)
#pragma unroll
        for (int ni = 0; ni < 4; ++ni)
#pragma unroll
            for (int j = 0; j < 8; ++j) {
                int r   = m0 + wm + mi * 16 + lhi * 8 + j;   // global M
                int col = n0 + wn + ni * 16 + lr;            // global N
                float f = acc[mi][ni][j];
                if constexpr (EPI == 0) {
                    float vb = f + bias[r];
                    unsigned short* q  = (unsigned short*)out0;
                    unsigned short* kk = (unsigned short*)out1;
                    unsigned short* vv = (unsigned short*)out2;
                    if (r < CCH)            q [(size_t)col * CCH + r]            = f2bf(vb * QK_SCALE);
                    else if (r < 2 * CCH)   kk[(size_t)col * CCH + (r - CCH)]    = f2bf(vb * QK_SCALE);
                    else                    vv[(size_t)(r - 2 * CCH) * TT + col] = f2bf(vb);
                } else if constexpr (EPI == 1) {
                    ((float*)out0)[(size_t)r * TT + col] = f;
                } else if constexpr (EPI == 2) {
                    ((unsigned short*)out0)[(size_t)r * CCH + col] = f2bf(f);
                } else {
                    ((float*)out0)[(size_t)r * TT + col] =
                        f + bias[r] + resid[(size_t)r * TT + col];
                }
            }
}

// ---------------------------------------------------------------------------
extern "C" void kernel_launch(void* const* d_in, const int* in_sizes, int n_in,
                              void* d_out, int out_size, void* d_ws, size_t ws_size,
                              hipStream_t stream) {
    const float* x     = (const float*)d_in[0];
    const float* gnw   = (const float*)d_in[1];
    const float* gnb   = (const float*)d_in[2];
    const float* qkvw  = (const float*)d_in[3];   // [1536][512]
    const float* qkvb  = (const float*)d_in[4];
    const float* projw = (const float*)d_in[5];   // [512][512]
    const float* projb = (const float*)d_in[6];
    float* out = (float*)d_out;

    const size_t MB = (size_t)1 << 20;
    char* ws = (char*)d_ws;
    float*          stats   = (float*)ws;                                   // 2 KB
    unsigned short* h       = (unsigned short*)(ws + 4096);                 // 4 MB  [T][C]
    unsigned short* qb      = (unsigned short*)(ws + 4096 + 4   * MB);      // 4 MB  [T][C]
    unsigned short* kb      = (unsigned short*)(ws + 4096 + 8   * MB);      // 4 MB  [T][C]
    unsigned short* vb      = (unsigned short*)(ws + 4096 + 12  * MB);      // 4 MB  [C][T]
    unsigned short* ot      = (unsigned short*)(ws + 4096 + 16  * MB);      // 4 MB  [T][C]
    float*          attn    = (float*)(ws + 4096 + 20 * MB);                // 64 MB [T][T]
    unsigned short* attn_bf = (unsigned short*)(ws + 4096 + 84 * MB);       // 32 MB [T][T]
    unsigned short* wq      = (unsigned short*)(ws + 4096 + 116 * MB);      // 1.5 MB
    unsigned short* wp      = (unsigned short*)(ws + 4096 + 118 * MB);      // 0.5 MB

    // weights -> bf16 (once)
    f32_to_bf16_kernel<<<(3 * CCH * CCH) / 256, 256, 0, stream>>>(qkvw, wq, 3 * CCH * CCH);
    f32_to_bf16_kernel<<<(CCH * CCH) / 256, 256, 0, stream>>>(projw, wp, CCH * CCH);

    gn_stats_kernel<<<BATCH * NGRP, 256, 0, stream>>>(x, stats);

    for (int b = 0; b < BATCH; ++b) {
        const float* xb = x + (size_t)b * CCH * TT;

        gn_apply_kernel<<<(CCH * TT) / 256, 256, 0, stream>>>(
            xb, stats + b * 2 * NGRP, gnw, gnb, h);

        // qkv = qkv_w @ h : M=1536, N=4096, K=512
        gemm128_wmma<0><<<dim3(12, 32), 256, 0, stream>>>(
            wq, CCH, h, CCH, CCH, qb, kb, vb, qkvb, nullptr);

        // attn = (q*s) @ (k*s)^T : M=N=4096, K=512
        gemm128_wmma<1><<<dim3(32, 32), 256, 0, stream>>>(
            qb, CCH, kb, CCH, CCH, attn, nullptr, nullptr, nullptr, nullptr);

        softmax_kernel<<<TT, 256, 0, stream>>>(attn, attn_bf);

        // o_t = attn @ v^T : M=4096, N=512, K=4096
        gemm128_wmma<2><<<dim3(32, 4), 256, 0, stream>>>(
            attn_bf, TT, vb, TT, TT, ot, nullptr, nullptr, nullptr, nullptr);

        // out = proj_w @ o_t^T + bias + x : M=512, N=4096, K=512
        gemm128_wmma<3><<<dim3(4, 32), 256, 0, stream>>>(
            wp, CCH, ot, CCH, CCH,
            out + (size_t)b * CCH * TT, nullptr, nullptr, projb, xb);
    }
}
